// GCLSTMCell_50483045597456
// MI455X (gfx1250) — matrix operand order repaired and checked by reading, not compiled, not run on hardware
//
#include <hip/hip_runtime.h>

typedef __bf16 bf16;
typedef _Float16 f16;
typedef __attribute__((ext_vector_type(16))) __bf16 v16bf;
typedef __attribute__((ext_vector_type(8)))  __bf16 v8bf;
typedef __attribute__((ext_vector_type(8)))  float  v8f;

__device__ __forceinline__ bf16 f2bf(float f) {
    unsigned u = __float_as_uint(f);
    u += 0x7FFFu + ((u >> 16) & 1u);          // round-to-nearest-even
    unsigned short h = (unsigned short)(u >> 16);
    bf16 r;
    __builtin_memcpy(&r, &h, 2);
    return r;
}

// ---------------- utility kernels ----------------
__global__ void fill_zero_kernel(float* p, long long n) {
    long long i = (long long)blockIdx.x * 256 + threadIdx.x;
    if (i < n) p[i] = 0.0f;
}

// Abuf = [x | h | 0 | -h]  (4 contiguous N*H blocks -> branch-free GEMM A addressing)
__global__ void init_a_kernel(float* __restrict__ A, const float* __restrict__ x,
                              const float* __restrict__ h, long long NH) {
    long long i = (long long)blockIdx.x * 256 + threadIdx.x;
    if (i < NH) {
        float hv = h[i];
        A[i]          = x[i];
        A[NH + i]     = hv;
        A[2 * NH + i] = 0.0f;
        A[3 * NH + i] = -hv;       // Tx2 seed: Tx2 = 2*L~*Tx1 - Tx0
    }
}

// deg[src[e]] += w[e]
__global__ void deg_kernel(const long long* __restrict__ ei,
                           const float* __restrict__ ew,
                           float* __restrict__ deg, int E) {
    int e = blockIdx.x * 256 + threadIdx.x;
    if (e < E) atomicAdd(&deg[ei[e]], ew[e]);
}

// deg -> dinv (in place)
__global__ void dinv_kernel(float* __restrict__ deg, int N) {
    int i = blockIdx.x * 256 + threadIdx.x;
    if (i < N) {
        float d = deg[i];
        deg[i] = (d > 0.0f) ? __frsqrt_rn(fmaxf(d, 1e-30f)) : 0.0f;
    }
}

// wn[e] = -w[e] * dinv[src] * dinv[dst]
__global__ void edge_norm_kernel(const long long* __restrict__ ei,
                                 const float* __restrict__ ew,
                                 const float* __restrict__ dinv,
                                 float* __restrict__ wn, int E) {
    int e = blockIdx.x * 256 + threadIdx.x;
    if (e < E) wn[e] = -ew[e] * dinv[ei[e]] * dinv[ei[(long long)E + e]];
}

// out[dst] += scale * wn[e] * v[src] ; one wave32 per edge, 4 feats/lane
__global__ __launch_bounds__(256) void spmm_kernel(
    const float* __restrict__ v, const float* __restrict__ wn,
    const long long* __restrict__ src, const long long* __restrict__ dst,
    float* __restrict__ out, float scale, int E) {
    long long t = (long long)blockIdx.x * 256 + threadIdx.x;
    int e = (int)(t >> 5);
    int lane = (int)(t & 31);
    if (e >= E) return;
    long long s = src[e], d = dst[e];
    float w = scale * wn[e];
    float4 val = *((const float4*)(v + s * 128) + lane);
    float* op = out + d * 128 + lane * 4;
    atomicAdd(op + 0, w * val.x);
    atomicAdd(op + 1, w * val.y);
    atomicAdd(op + 2, w * val.z);
    atomicAdd(op + 3, w * val.w);
}

// Pack B^T (n-major, 512x512 bf16): Bt[n][k], n = g*128+j,
// k<128 -> W_g[k][j]; k>=128 -> theta_g[(k-128)*128 + j] (theta contiguous (3,128,128))
__global__ void pack_b_kernel(const float* __restrict__ W_i, const float* __restrict__ W_f,
                              const float* __restrict__ W_c, const float* __restrict__ W_o,
                              const float* __restrict__ th_i, const float* __restrict__ th_f,
                              const float* __restrict__ th_c, const float* __restrict__ th_o,
                              bf16* __restrict__ Bt) {
    int t = blockIdx.x * 256 + threadIdx.x;      // 0..262143
    int n = t >> 9, k = t & 511;
    int g = n >> 7, j = n & 127;
    const float* W  = (g == 0) ? W_i  : (g == 1) ? W_f  : (g == 2) ? W_c  : W_o;
    const float* TH = (g == 0) ? th_i : (g == 1) ? th_f : (g == 2) ? th_c : th_o;
    float v = (k < 128) ? W[k * 128 + j] : TH[(k - 128) * 128 + j];
    Bt[(long long)n * 512 + k] = f2bf(v);
}

// ---------------- WMMA fragment loaders (ISA 7.12.2 layouts) ----------------
// A (16x32 bf16): lane<16 row m=lane, K in {0..7}U{16..23}; lane>=16 K in {8..15}U{24..31}
__device__ __forceinline__ v16bf load_frag_a(const bf16* base, int lane) {
    int m = lane & 15;
    int koff = (lane < 16) ? 0 : 8;
    const bf16* p = base + m * 32 + koff;
    v8bf lo = *(const v8bf*)(p);
    v8bf hi = *(const v8bf*)(p + 16);
    return __builtin_shufflevector(lo, hi, 0, 1, 2, 3, 4, 5, 6, 7,
                                   8, 9, 10, 11, 12, 13, 14, 15);
}
// B (32x16 bf16), stored transposed in LDS (n-major, stride 32):
// lane<16 col n=nbase+lane, K in {0..7}U{16..23}; lane>=16 K in {8..15}U{24..31}
__device__ __forceinline__ v16bf load_frag_b(const bf16* base, int nbase, int lane) {
    int n = nbase + (lane & 15);
    int koff = (lane < 16) ? 0 : 8;
    const bf16* p = base + n * 32 + koff;
    v8bf lo = *(const v8bf*)(p);
    v8bf hi = *(const v8bf*)(p + 16);
    return __builtin_shufflevector(lo, hi, 0, 1, 2, 3, 4, 5, 6, 7,
                                   8, 9, 10, 11, 12, 13, 14, 15);
}

// ---------------- fused GEMM (N x 512 x 512, bf16 WMMA) + LSTM epilogue ----------------
// Block: 256 threads (8 waves), 32 rows x 512 gate-cols per block.
// Wave w: rows [16*(w>>2), +16), cols [(w&3)*128, +128) -> 8 wmma accumulators.
// LDS: 32KB B tile (aliased by f16 gate buffer after the loop) + 2KB A tile = 34.8KB.
__global__ __launch_bounds__(256) void gclstm_gemm_kernel(
    const float* __restrict__ Abuf, long long NH,
    const bf16* __restrict__ Bt, const float* __restrict__ c_prev,
    const float* __restrict__ b_i, const float* __restrict__ b_f,
    const float* __restrict__ b_c, const float* __restrict__ b_o,
    const float* __restrict__ w_ci, const float* __restrict__ w_cf,
    const float* __restrict__ w_co,
    float* __restrict__ out_h, float* __restrict__ out_c, int n_rows) {

    __shared__ __align__(16) unsigned char smem[512 * 32 * 2]; // B tile / f16 gates (32KB)
    __shared__ __align__(16) bf16 a_sh[32 * 32];               // A tile (2KB)

    const int tid = threadIdx.x;
    const int lane = tid & 31;
    const int w = tid >> 5;
    const int rhalf = w >> 2;            // which 16-row half of the 32-row block tile
    const int cbase = (w & 3) * 128;     // which gate's 128 columns
    const long long row0 = (long long)blockIdx.x * 32;

    const uint4* bg = (const uint4*)Bt;  // Bt row = 64 uint4
    const int ar = tid >> 3, ac4 = (tid & 7) * 4;
    const long long agrow = row0 + ar;
    const bool arow_ok = (agrow < n_rows);

    v8f acc[8];
#pragma unroll
    for (int i = 0; i < 8; ++i)
#pragma unroll
        for (int j = 0; j < 8; ++j) acc[i][j] = 0.0f;

#pragma unroll
    for (int cc = 0; cc < 16; ++cc) {    // K = 512 in 16 chunks of 32, branch-free addressing
        const float* src = Abuf + (long long)(cc >> 2) * NH + (cc & 3) * 32;

        __syncthreads();
        // stage A chunk (32 rows x 32 k) fp32 -> bf16
        {
            float4 v = make_float4(0.0f, 0.0f, 0.0f, 0.0f);
            if (arow_ok) v = *(const float4*)(src + agrow * 128 + ac4);
            bf16* ad = a_sh + ar * 32 + ac4;
            ad[0] = f2bf(v.x); ad[1] = f2bf(v.y);
            ad[2] = f2bf(v.z); ad[3] = f2bf(v.w);
        }
        // stage B chunk (512 n x 32 k bf16) = 2048 uint4
        {
            uint4* bd = (uint4*)smem;
#pragma unroll
            for (int i = 0; i < 8; ++i) {
                int q = tid + 256 * i;
                bd[q] = bg[(q >> 2) * 64 + cc * 4 + (q & 3)];
            }
        }
        __syncthreads();

        v16bf fa = load_frag_a(a_sh + rhalf * 16 * 32, lane);
        const bf16* bb = (const bf16*)smem;
#pragma unroll
        for (int n0 = 0; n0 < 8; ++n0) {
            v16bf fb = load_frag_b(bb, cbase + n0 * 16, lane);
            acc[n0] = __builtin_amdgcn_wmma_f32_16x16x32_bf16(
                false, fa, false, fb, (short)0, acc[n0], false, false);
        }
    }

    __syncthreads();
    // D layout: lane<16 -> N=lane, VGPR r -> M=r ; lane>=16 -> N=lane-16, M=r+8
    f16* gf = (f16*)smem;                 // 32 rows x 512 cols, f16 (exactly 32KB)
    {
        int moff = (lane < 16) ? 0 : 8;
        int nloc = lane & 15;
#pragma unroll
        for (int n0 = 0; n0 < 8; ++n0) {
            int col = cbase + n0 * 16 + nloc;
#pragma unroll
            for (int r = 0; r < 8; ++r)
                gf[(rhalf * 16 + r + moff) * 512 + col] = (f16)acc[n0][r];
        }
    }
    __syncthreads();

    // fused LSTM pointwise epilogue: 32 rows x 128 features
#pragma unroll
    for (int p = tid; p < 32 * 128; p += 256) {
        int r = p >> 7, j = p & 127;
        long long grow = row0 + r;
        if (grow >= n_rows) continue;
        float cp = c_prev[grow * 128 + j];
        float pi = (float)gf[r * 512 + j]       + b_i[j] + w_ci[j] * cp;
        float pf = (float)gf[r * 512 + 128 + j] + b_f[j] + w_cf[j] * cp;
        float pc = (float)gf[r * 512 + 256 + j] + b_c[j];
        float po = (float)gf[r * 512 + 384 + j] + b_o[j];
        float ig = 1.0f / (1.0f + __expf(-pi));
        float fg = 1.0f / (1.0f + __expf(-pf));
        float tg = tanhf(pc);
        float ct = fg * cp + ig * tg;
        float og = 1.0f / (1.0f + __expf(-(po + w_co[j] * ct)));
        float ht = og * tanhf(ct);
        out_h[grow * 128 + j] = ht;
        out_c[grow * 128 + j] = ct;
    }
}

extern "C" void kernel_launch(void* const* d_in, const int* in_sizes, int n_in,
                              void* d_out, int out_size, void* d_ws, size_t ws_size,
                              hipStream_t stream) {
    const float* x_t    = (const float*)d_in[0];
    const float* h_prev = (const float*)d_in[1];
    const float* c_prev = (const float*)d_in[2];
    const float* ew     = (const float*)d_in[3];
    const long long* ei = (const long long*)d_in[4];
    const float* W_i = (const float*)d_in[5];
    const float* W_f = (const float*)d_in[6];
    const float* W_c = (const float*)d_in[7];
    const float* W_o = (const float*)d_in[8];
    const float* th_i = (const float*)d_in[9];
    const float* th_f = (const float*)d_in[10];
    const float* th_c = (const float*)d_in[11];
    const float* th_o = (const float*)d_in[12];
    const float* b_i = (const float*)d_in[13];
    const float* b_f = (const float*)d_in[14];
    const float* b_c = (const float*)d_in[15];
    const float* b_o = (const float*)d_in[16];
    const float* w_ci = (const float*)d_in[17];
    const float* w_cf = (const float*)d_in[18];
    const float* w_co = (const float*)d_in[19];

    const int H = 128;
    const int N = in_sizes[2] / H;      // c_prev is (N, H)
    const int E = in_sizes[3];          // edge_weight is (E,)
    const long long NH = (long long)N * H;

    // workspace layout: Abuf = [x | h | Tx1 | Tx2] (4*NH f32), then deg, wn, packed B
    const long long Np = ((long long)N + 63) & ~63LL;
    float* Abuf = (float*)d_ws;            // 4 * NH
    float* tx1  = Abuf + 2 * NH;           // alias inside Abuf
    float* tx2  = Abuf + 3 * NH;           // alias inside Abuf
    float* deg  = Abuf + 4 * NH;           // N (becomes dinv in place)
    float* wn   = deg + Np;                // E
    bf16*  Bt   = (bf16*)(wn + E);         // 512*512 bf16 (transposed, n-major)

    float* out_h = (float*)d_out;
    float* out_c = out_h + NH;

    const int T = 256;
    // init
    init_a_kernel<<<(int)((NH + T - 1) / T), T, 0, stream>>>(Abuf, x_t, h_prev, NH);
    fill_zero_kernel<<<(int)((N + T - 1) / T), T, 0, stream>>>(deg, N);
    pack_b_kernel<<<(512 * 512) / T, T, 0, stream>>>(W_i, W_f, W_c, W_o,
                                                     th_i, th_f, th_c, th_o, Bt);
    // graph normalization
    deg_kernel<<<(E + T - 1) / T, T, 0, stream>>>(ei, ew, deg, E);
    dinv_kernel<<<(N + T - 1) / T, T, 0, stream>>>(deg, N);
    edge_norm_kernel<<<(E + T - 1) / T, T, 0, stream>>>(ei, ew, deg, wn, E);
    // Chebyshev recursion: Tx1 = L~ h ; Tx2 += 2 L~ Tx1 (Tx2 seeded with -h)
    long long spmm_threads = (long long)E * 32;
    int spmm_blocks = (int)((spmm_threads + T - 1) / T);
    spmm_kernel<<<spmm_blocks, T, 0, stream>>>(h_prev, wn, ei, ei + E, tx1, 1.0f, E);
    spmm_kernel<<<spmm_blocks, T, 0, stream>>>(tx1, wn, ei, ei + E, tx2, 2.0f, E);
    // fused bf16-WMMA GEMM (N x 512 x 512) + LSTM epilogue
    int gemm_blocks = (N + 31) / 32;
    gclstm_gemm_kernel<<<gemm_blocks, T, 0, stream>>>(
        Abuf, NH, Bt, c_prev,
        b_i, b_f, b_c, b_o, w_ci, w_cf, w_co,
        out_h, out_c, N);
}
